// _ligru_cell_jit_68942815035797
// MI455X (gfx1250) — matrix operand order repaired and checked by reading, not compiled, not run on hardware
//
#include <hip/hip_runtime.h>

// ---------------------------------------------------------------------------
// LiGRU cell, persistent single-workgroup recurrence kernel for gfx1250.
//   B=32, T=1000, H=512, 2H=1024.
//   GEMM per step: [32,512] x [512,1024] via v_wmma_f32_16x16x32_f16.
// ---------------------------------------------------------------------------

typedef __attribute__((ext_vector_type(16))) _Float16 v16h;
typedef __attribute__((ext_vector_type(8)))  float    v8f;
typedef __attribute__((ext_vector_type(4)))  unsigned int v4u;

#define B_    32
#define T_    1000
#define H_    512
#define TWOH  1024
#define KDIM  512
#define H16S  520    // padded f16 row stride for LDS h16 (1040 B, 16B-aligned)
#define YS    1028   // padded f32 row stride for LDS ybuf (bank-conflict-free)

// LDS carve (bytes)
#define YBUF_OFF  0
#define YBUF_SZ   (32 * YS * 4)            // 131584
#define H32_OFF   (YBUF_OFF + YBUF_SZ)     // 131584
#define H32_SZ    (32 * H_ * 4)            // 65536
#define H16_OFF   (H32_OFF + H32_SZ)       // 197120 (16B aligned)
#define H16_SZ    (32 * H16S * 2)          // 33280
#define GAM_OFF   (H16_OFF + H16_SZ)       // 230400
#define BET_OFF   (GAM_OFF + 4096)         // 234496
#define LDS_TOTAL (BET_OFF + 4096)         // 238592  (< 320 KB / WGP)

// u (fp32 [1024][512]) -> f16, done once per launch into d_ws (L2-resident).
__global__ void convert_u_f16(const float* __restrict__ u,
                              _Float16* __restrict__ u16, int n) {
    int i = blockIdx.x * blockDim.x + threadIdx.x;
    if (i < n) u16[i] = (_Float16)u[i];
}

// Load a 32-byte WMMA fragment as two 16-byte chunks at p and p+elemOff.
__device__ __forceinline__ v16h load_frag(const _Float16* p, int elemOff) {
    v16h f;
    v4u* d = reinterpret_cast<v4u*>(&f);
    d[0] = *reinterpret_cast<const v4u*>(p);
    d[1] = *reinterpret_cast<const v4u*>(p + elemOff);
    return f;
}

__global__ __launch_bounds__(1024, 1)
void ligru_persistent(const float* __restrict__ wx,       // [B,T,2H]
                      const _Float16* __restrict__ u16,   // [2H,H] f16
                      const float* __restrict__ ht,       // [B,H]
                      const float* __restrict__ dm,       // [B,H]
                      const float* __restrict__ gamma_,   // [2H]
                      const float* __restrict__ beta_,    // [2H]
                      float* __restrict__ out)
{
    extern __shared__ char smem[];
    float*    ybuf = reinterpret_cast<float*>(smem + YBUF_OFF);    // [32][YS]
    float*    h32  = reinterpret_cast<float*>(smem + H32_OFF);     // [32][512]
    _Float16* h16  = reinterpret_cast<_Float16*>(smem + H16_OFF);  // [32][H16S]
    float*    gsh  = reinterpret_cast<float*>(smem + GAM_OFF);     // [1024]
    float*    bsh  = reinterpret_cast<float*>(smem + BET_OFF);     // [1024]

    const int tid = threadIdx.x;
    const int l   = tid & 31;    // lane within wave32
    const int w   = tid >> 5;    // wave id 0..31
    const int lm  = l & 15;
    const int kh  = l >> 4;

    // ---- init state + constants ----
    for (int i = tid; i < B_ * H_; i += 1024) {
        int b = i >> 9, j = i & (H_ - 1);
        float v = ht[i];
        h32[b * H_ + j]   = v;
        h16[b * H16S + j] = (_Float16)v;
    }
    for (int i = tid; i < TWOH; i += 1024) { gsh[i] = gamma_[i]; bsh[i] = beta_[i]; }

    const size_t BTH = (size_t)B_ * T_ * H_;
    if (tid == 0) out[4 * BTH] = 1e-5f;   // EPS scalar output

    float* out_h  = out;
    float* out_z  = out + BTH;
    float* out_a  = out + 2 * BTH;
    float* out_hc = out + 3 * BTH;
    float* out_xh = out + 4 * BTH + 1;

    // A-fragment lane layout (16-bit, 16x32): lanes 0-15 K=0-7 & 16-23,
    // lanes 16-31 K=8-15 & 24-31 -> chunks at kbase and kbase+16.
    const _Float16* a0base = h16 + lm * H16S + kh * 8;
    const _Float16* a1base = a0base + 16 * H16S;
    // B-fragment (32x16): same K order as A: lanes 0-15 K=0-7 & 16-23,
    // lanes 16-31 K=8-15 & 24-31 -> chunks at kbase and kbase+16. N = lane%16.
    const int nt0 = 2 * w;                         // this wave's N tiles
    const _Float16* b0base = u16 + (size_t)(nt0 * 16 + lm) * KDIM + kh * 8;
    const _Float16* b1base = b0base + 16 * KDIM;   // N tile nt0+1

    for (int t = 0; t < T_; ++t) {
        __syncthreads();   // h16 from previous row-phase visible to all waves

        // Opaque zero offset: stops the compiler from proving the B-fragment
        // global loads loop-invariant across t (would hoist + spill).
        int toff = 0;
        asm volatile("" : "+v"(toff));
        const _Float16* bp0 = b0base + toff;
        const _Float16* bp1 = b1base + toff;

        // ---------------- GEMM phase: y = h @ u.T ----------------
        // unroll capped at 2: bounds in-flight fragments so the 8-wave/SIMD
        // VGPR budget (~128) is never exceeded (no scratch spills).
        v8f acc00 = {}, acc01 = {}, acc10 = {}, acc11 = {};
        #pragma unroll 2
        for (int kk = 0; kk < 16; ++kk) {
            const int ko = kk * 32;
            v16h A0 = load_frag(a0base + ko, 16);  // LDS: ds_load_b128 x2
            v16h A1 = load_frag(a1base + ko, 16);
            v16h B0 = load_frag(bp0 + ko, 16);     // L2:  global_load_b128 x2
            v16h B1 = load_frag(bp1 + ko, 16);
            acc00 = __builtin_amdgcn_wmma_f32_16x16x32_f16(false, A0, false, B0,
                        (short)0, acc00, false, false);
            acc01 = __builtin_amdgcn_wmma_f32_16x16x32_f16(false, A0, false, B1,
                        (short)0, acc01, false, false);
            acc10 = __builtin_amdgcn_wmma_f32_16x16x32_f16(false, A1, false, B0,
                        (short)0, acc10, false, false);
            acc11 = __builtin_amdgcn_wmma_f32_16x16x32_f16(false, A1, false, B1,
                        (short)0, acc11, false, false);
        }
        // C/D layout: VGPR r -> lanes 0-15 M=r, lanes 16-31 M=r+8, N=lane%16.
        {
            const int n0 = nt0 * 16 + lm;
            const int n1 = n0 + 16;
            #pragma unroll
            for (int r = 0; r < 8; ++r) {
                int m = r + 8 * kh;
                ybuf[m * YS + n0]        = acc00[r];
                ybuf[m * YS + n1]        = acc01[r];
                ybuf[(16 + m) * YS + n0] = acc10[r];
                ybuf[(16 + m) * YS + n1] = acc11[r];
            }
        }
        __syncthreads();

        // ------------- Row phase: wave w owns batch row b = w -------------
        {
            const int b = w;
            const float* wxrow = wx + ((size_t)b * T_ + t) * TWOH;

            // LayerNorm stats over 1024 cols (32 per lane, wave reduction)
            float yv[32];
            float s = 0.f, ss = 0.f;
            #pragma unroll
            for (int i = 0; i < 32; ++i) {
                float v = ybuf[b * YS + i * 32 + l];
                yv[i] = v; s += v; ss += v * v;
            }
            #pragma unroll
            for (int off = 16; off > 0; off >>= 1) {
                s  += __shfl_xor(s,  off, 32);
                ss += __shfl_xor(ss, off, 32);
            }
            const float mu  = s * (1.f / 1024.f);
            const float var = ss * (1.f / 1024.f) - mu * mu;
            const float rs  = rsqrtf(var + 1e-5f);

            // x_hat, gates; stash gates back into ybuf row.
            // wx is read exactly once -> nontemporal (keep L2 for u16);
            // outputs are write-once streams -> nontemporal stores.
            const size_t xo = ((size_t)b * T_ + t) * TWOH;
            #pragma unroll 4
            for (int i = 0; i < 32; ++i) {
                int idx = i * 32 + l;
                float xh = (yv[i] - mu) * rs * gsh[idx] + bsh[idx];
                float g  = __builtin_nontemporal_load(wxrow + idx) + xh;
                ybuf[b * YS + idx] = g;
                __builtin_nontemporal_store(xh, out_xh + xo + idx);
            }

            // gating + h update (at j, zt j+512: both written by this lane)
            const size_t ho = ((size_t)b * T_ + t) * H_;
            #pragma unroll 4
            for (int i = 0; i < 16; ++i) {
                int j = i * 32 + l;
                float a  = ybuf[b * YS + j];
                float zr = ybuf[b * YS + 512 + j];
                float z  = 1.0f / (1.0f + __expf(-zr));
                float hc = fmaxf(a, 0.f) * dm[b * H_ + j];
                float hp = h32[b * H_ + j];
                float hn = hp * z + (1.f - z) * hc;
                h32[b * H_ + j]   = hn;
                h16[b * H16S + j] = (_Float16)hn;
                __builtin_nontemporal_store(hn, out_h  + ho + j);
                __builtin_nontemporal_store(z,  out_z  + ho + j);
                __builtin_nontemporal_store(a,  out_a  + ho + j);
                __builtin_nontemporal_store(hc, out_hc + ho + j);
            }

            // prefetch next step's wx row into cache (global_prefetch_b8)
            if (t + 1 < T_) {
                __builtin_prefetch(wxrow + TWOH + l * 32, 0, 1);
            }
        }
    }
}

extern "C" void kernel_launch(void* const* d_in, const int* in_sizes, int n_in,
                              void* d_out, int out_size, void* d_ws, size_t ws_size,
                              hipStream_t stream) {
    (void)in_sizes; (void)n_in; (void)out_size; (void)ws_size;
    const float* wx    = (const float*)d_in[0];
    const float* u     = (const float*)d_in[1];
    const float* ht    = (const float*)d_in[2];
    const float* dm    = (const float*)d_in[3];
    const float* gamma = (const float*)d_in[4];
    const float* beta  = (const float*)d_in[5];
    float* out = (float*)d_out;

    _Float16* u16 = (_Float16*)d_ws;   // 1 MB scratch for f16 weights
    const int n = TWOH * KDIM;
    convert_u_f16<<<(n + 255) / 256, 256, 0, stream>>>(u, u16, n);

    ligru_persistent<<<1, 1024, LDS_TOTAL, stream>>>(wx, u16, ht, dm,
                                                     gamma, beta, out);
}
